// MultiHeadAttention_15418932592942
// MI455X (gfx1250) — matrix operand order repaired and checked by reading, not compiled, not run on hardware
//
#include <hip/hip_runtime.h>
#include <hip/hip_bf16.h>
#include <cstdint>

typedef __bf16 bf16_t;
typedef __attribute__((ext_vector_type(16))) __bf16 v16bf;
typedef __attribute__((ext_vector_type(8)))  __bf16 v8bf;
typedef __attribute__((ext_vector_type(8)))  float  v8f;
typedef __attribute__((ext_vector_type(4)))  float  v4f;

static constexpr int D_MODEL = 512;
static constexpr int SEQ     = 2048;
static constexpr int NB      = 4;
static constexpr int NH      = 8;
static constexpr int DH      = 64;

// ---------------------------------------------------------------------------
// Fragment helpers (CDNA5 WMMA 16x16x32 bf16 layouts, CDNA5 ISA §7.12.2)
// A (16x32, MxK): lane row = lane%16; lanes 0-15: K={0..7,16..23},
//                 lanes 16-31: K={8..15,24..31}; sequential within 8-chunks.
// B (32x16, KxN): lane col = lane%16; lanes 0-15: K={0..7,16..23}, lanes 16-31:
//                 K={8..15,24..31} (same as A), read from N-major storage.
// C/D (16x16 f32): lane col = lane%16, VGPR v holds row v + (lane<16?0:8).
// ---------------------------------------------------------------------------

static __device__ __forceinline__ v16bf frag_cat(v8bf lo, v8bf hi8) {
  v16bf r;
#pragma unroll
  for (int i = 0; i < 8; ++i) { r[i] = lo[i]; r[8 + i] = hi8[i]; }
  return r;
}

static __device__ __forceinline__ v16bf load_a_frag(const bf16_t* row, int hi) {
  const int koff = hi * 8;
  v8bf lo = *(const v8bf*)(row + koff);
  v8bf up = *(const v8bf*)(row + koff + 16);
  return frag_cat(lo, up);
}

static __device__ __forceinline__ v16bf load_b_frag(const bf16_t* row, int hi) {
  const int koff = hi * 8;
  v8bf lo = *(const v8bf*)(row + koff);
  v8bf up = *(const v8bf*)(row + koff + 16);
  return frag_cat(lo, up);
}

static __device__ __forceinline__ void cvt8(const float* p, v16bf& a, int base) {
  v4f x0 = *(const v4f*)(p);
  v4f x1 = *(const v4f*)(p + 4);
#pragma unroll
  for (int i = 0; i < 4; ++i) {
    a[base + i]     = (bf16_t)x0[i];
    a[base + 4 + i] = (bf16_t)x1[i];
  }
}

static __device__ __forceinline__ v8f wmma_bf16(v16bf a, v16bf b, v8f c) {
  return __builtin_amdgcn_wmma_f32_16x16x32_bf16(false, a, false, b,
                                                 (short)0, c, false, false);
}

// Async global->LDS copy of one 16B chunk per lane (tracked by ASYNCcnt).
static __device__ __forceinline__ void async_copy_b128(uint32_t lds_off,
                                                       uint64_t sbase,
                                                       uint32_t goff) {
  asm volatile("global_load_async_to_lds_b128 %0, %1, %2"
               :: "v"(lds_off), "v"(goff), "s"(sbase)
               : "memory");
}
static __device__ __forceinline__ void wait_async0() {
  asm volatile("s_wait_asynccnt 0x0" ::: "memory");
}
static __device__ __forceinline__ void wait_ds0() {
  asm volatile("s_wait_dscnt 0x0" ::: "memory");
}

// ---------------------------------------------------------------------------
// Kernel 1: fused Q/K/V projection  y = x @ W^T + b, bf16 output.
//   qp, kp : [B, H, S, Dh]; vpT : [B, H, Dh, S] (transposed for P·V B-frags)
// grid (128, 8, 3), block 128 (4 waves, each 16 rows x 64 cols)
// ---------------------------------------------------------------------------
__global__ __launch_bounds__(128)
void proj_kernel(const float* __restrict__ xq, const float* __restrict__ xk,
                 const float* __restrict__ xv,
                 const float* __restrict__ wq, const float* __restrict__ wk,
                 const float* __restrict__ wv,
                 const float* __restrict__ bq, const float* __restrict__ bk,
                 const float* __restrict__ bv,
                 bf16_t* __restrict__ qp, bf16_t* __restrict__ kp,
                 bf16_t* __restrict__ vpT) {
  const int sel = blockIdx.z;
  const float* x    = (sel == 0) ? xq : (sel == 1) ? xk : xv;
  const float* w    = (sel == 0) ? wq : (sel == 1) ? wk : wv;
  const float* bias = (sel == 0) ? bq : (sel == 1) ? bk : bv;
  bf16_t* dst = (sel == 0) ? qp : kp;

  const int wave = threadIdx.x >> 5;
  const int lane = threadIdx.x & 31;
  const int lr = lane & 15;
  const int hi = lane >> 4;

  const int m0 = blockIdx.x * 64 + wave * 16;
  const int n0 = blockIdx.y * 64;
  const int arow = m0 + lr;

  v8f acc[4] = {{}, {}, {}, {}};

  for (int kk = 0; kk < D_MODEL; kk += 32) {
    const float* ap = x + arow * D_MODEL + kk + hi * 8;
    v16bf a;
    cvt8(ap, a, 0);
    cvt8(ap + 16, a, 8);
#pragma unroll
    for (int jj = 0; jj < 4; ++jj) {
      const int e = n0 + jj * 16 + lr;
      const float* bp = w + e * D_MODEL + kk + hi * 8;  // W is N-major (B=W^T)
      v16bf b;
      cvt8(bp, b, 0);
      cvt8(bp + 16, b, 8);
      acc[jj] = wmma_bf16(a, b, acc[jj]);
    }
  }

#pragma unroll
  for (int jj = 0; jj < 4; ++jj) {
    const int e = n0 + jj * 16 + lr;
    const float be = bias[e];
    const int h = e >> 6, dh = e & (DH - 1);
#pragma unroll
    for (int v = 0; v < 8; ++v) {
      const int m = m0 + hi * 8 + v;
      const int b = m >> 11, s = m & (SEQ - 1);
      const float val = acc[jj][v] + be;
      if (sel < 2)
        dst[(((b * NH + h) * SEQ) + s) * DH + dh] = (bf16_t)val;
      else
        vpT[(((b * NH + h) * DH) + dh) * SEQ + s] = (bf16_t)val;
    }
  }
}

// ---------------------------------------------------------------------------
// Kernel 2: flash attention with double-buffered async K/V staging in LDS.
// Block = 4 waves (64 q rows, 16 per wave); 32-key tiles; online softmax.
// grid (S/64=32, H, B), block 128.
// ---------------------------------------------------------------------------
__global__ __launch_bounds__(128)
void attn_kernel(const bf16_t* __restrict__ qp, const bf16_t* __restrict__ kp,
                 const bf16_t* __restrict__ vpT, float* __restrict__ out) {
  // K tile: 32 keys x 64 dh, padded to 72 (144B rows -> conflict-free frags)
  // V tile: 64 dh  x 32 keys, padded to 40 (80B rows)
  __shared__ alignas(16) bf16_t kbuf[2][32][72];
  __shared__ alignas(16) bf16_t vbuf[2][64][40];
  __shared__ alignas(16) bf16_t pstage[4][16][40];  // per-wave P transpose

  const int tid = threadIdx.x;
  const int wave = tid >> 5;
  const int lane = tid & 31;
  const int lr = lane & 15;
  const int hi = lane >> 4;

  const int b = blockIdx.z, h = blockIdx.y;
  const int q0 = blockIdx.x * 64 + wave * 16;

  const bf16_t* qbh = qp + ((b * NH + h) * SEQ) * DH;
  const uint64_t kbase = (uint64_t)(uintptr_t)(kp + ((b * NH + h) * SEQ) * DH);
  const uint64_t vbase = (uint64_t)(uintptr_t)(vpT + ((b * NH + h) * DH) * SEQ);

  // Cooperative async staging of one 32-key K tile + V tile (8KB / block).
  auto stage_tiles = [&](int buf, int kt) {
#pragma unroll
    for (int i = 0; i < 2; ++i) {  // K: 256 x 16B chunks
      const int chunk = tid + i * 128;
      const int row = chunk >> 3, seg = chunk & 7;
      const uint32_t lds = (uint32_t)(uintptr_t)(&kbuf[buf][row][seg * 8]);
      const uint32_t gof = (uint32_t)(((kt + row) * DH + seg * 8) * 2);
      async_copy_b128(lds, kbase, gof);
    }
#pragma unroll
    for (int i = 0; i < 2; ++i) {  // V: 256 x 16B chunks
      const int chunk = tid + i * 128;
      const int row = chunk >> 2, seg = chunk & 3;
      const uint32_t lds = (uint32_t)(uintptr_t)(&vbuf[buf][row][seg * 8]);
      const uint32_t gof = (uint32_t)((row * SEQ + kt + seg * 8) * 2);
      async_copy_b128(lds, vbase, gof);
    }
  };

  // Q fragments (Dh=64 -> two K=32 chunks), loaded once.
  const bf16_t* qrow = qbh + (q0 + lr) * DH;
  v16bf qa[2];
#pragma unroll
  for (int c = 0; c < 2; ++c) qa[c] = load_a_frag(qrow + c * 32, hi);

  float mrow[8], lrow[8];
  v8f ctx[4] = {{}, {}, {}, {}};
#pragma unroll
  for (int v = 0; v < 8; ++v) { mrow[v] = -1.0e30f; lrow[v] = 0.0f; }

  stage_tiles(0, 0);
  wait_async0();
  __syncthreads();

  int buf = 0;
  for (int kt = 0; kt < SEQ; kt += 32, buf ^= 1) {
    if (kt + 32 < SEQ) stage_tiles(buf ^ 1, kt + 32);  // prefetch next tile

    // ---- scores S = (Q · K^T) * 1/sqrt(Dh): two 16x16 tiles -------------
    v8f sc[2];
#pragma unroll
    for (int j = 0; j < 2; ++j) {
      v8f c = {};
#pragma unroll
      for (int cc = 0; cc < 2; ++cc) {
        const bf16_t* krow = &kbuf[buf][j * 16 + lr][cc * 32];
        c = wmma_bf16(qa[cc], load_b_frag(krow, hi), c);
      }
      sc[j] = c;
    }

    // ---- online softmax --------------------------------------------------
    float newmax[8];
#pragma unroll
    for (int v = 0; v < 8; ++v) {
      const float s0 = sc[0][v] * 0.125f;
      const float s1 = sc[1][v] * 0.125f;
      sc[0][v] = s0; sc[1][v] = s1;
      float mx = fmaxf(s0, s1);
      mx = fmaxf(mx, __shfl_xor(mx, 1, 16));
      mx = fmaxf(mx, __shfl_xor(mx, 2, 16));
      mx = fmaxf(mx, __shfl_xor(mx, 4, 16));
      mx = fmaxf(mx, __shfl_xor(mx, 8, 16));
      newmax[v] = fmaxf(mrow[v], mx);
    }
#pragma unroll
    for (int v = 0; v < 8; ++v) {
      const float corr = __expf(mrow[v] - newmax[v]);
      mrow[v] = newmax[v];
      const float p0 = __expf(sc[0][v] - newmax[v]);
      const float p1 = __expf(sc[1][v] - newmax[v]);
      sc[0][v] = p0; sc[1][v] = p1;
      float ps = p0 + p1;
      ps += __shfl_xor(ps, 1, 16);
      ps += __shfl_xor(ps, 2, 16);
      ps += __shfl_xor(ps, 4, 16);
      ps += __shfl_xor(ps, 8, 16);
      lrow[v] = lrow[v] * corr + ps;
#pragma unroll
      for (int d = 0; d < 4; ++d) ctx[d][v] *= corr;
    }

    // ---- transpose P (C/D -> A layout) via wave-private LDS slab --------
#pragma unroll
    for (int j = 0; j < 2; ++j)
#pragma unroll
      for (int v = 0; v < 8; ++v)
        pstage[wave][v + hi * 8][j * 16 + lr] = (bf16_t)sc[j][v];
    wait_ds0();  // wave-local ordering; no block barrier needed
    const v16bf pa = load_a_frag(&pstage[wave][lr][0], hi);

    // ---- ctx += P · V ----------------------------------------------------
#pragma unroll
    for (int d = 0; d < 4; ++d) {
      const bf16_t* vrow = &vbuf[buf][d * 16 + lr][0];
      ctx[d] = wmma_bf16(pa, load_b_frag(vrow, hi), ctx[d]);
    }

    wait_async0();    // next tile's DMA complete (this wave's share)
    __syncthreads();  // all waves done reading cur + staging next
  }

  // ---- normalize and store f32 output [B, S, H*Dh] ----------------------
#pragma unroll
  for (int v = 0; v < 8; ++v) {
    const float inv = 1.0f / lrow[v];
    const int qr = q0 + hi * 8 + v;
#pragma unroll
    for (int d = 0; d < 4; ++d) {
      const int dh = d * 16 + lr;
      out[(b * SEQ + qr) * D_MODEL + h * DH + dh] = ctx[d][v] * inv;
    }
  }
}

// ---------------------------------------------------------------------------
extern "C" void kernel_launch(void* const* d_in, const int* in_sizes, int n_in,
                              void* d_out, int out_size, void* d_ws,
                              size_t ws_size, hipStream_t stream) {
  (void)in_sizes; (void)n_in; (void)out_size; (void)ws_size;
  const float* q  = (const float*)d_in[0];
  const float* k  = (const float*)d_in[1];
  const float* v  = (const float*)d_in[2];
  const float* wq = (const float*)d_in[3];
  const float* bq = (const float*)d_in[4];
  const float* wk = (const float*)d_in[5];
  const float* bk = (const float*)d_in[6];
  const float* wv = (const float*)d_in[7];
  const float* bv = (const float*)d_in[8];

  const size_t PELEMS = (size_t)NB * SEQ * D_MODEL;
  bf16_t* qp  = (bf16_t*)d_ws;
  bf16_t* kp  = qp + PELEMS;
  bf16_t* vpT = kp + PELEMS;

  dim3 pgrid((NB * SEQ) / 64, D_MODEL / 64, 3);
  proj_kernel<<<pgrid, dim3(128), 0, stream>>>(q, k, v, wq, wk, wv, bq, bk, bv,
                                               qp, kp, vpT);

  dim3 agrid(SEQ / 64, NH, NB);
  attn_kernel<<<agrid, dim3(128), 0, stream>>>(qp, kp, vpT, (float*)d_out);
}
